// ResGatedMultiDiGraphNet_3908420240096
// MI455X (gfx1250) — matrix operand order repaired and checked by reading, not compiled, not run on hardware
//
#include <hip/hip_runtime.h>

#define H 128

typedef __attribute__((ext_vector_type(16))) __bf16 v16bf;
typedef __attribute__((ext_vector_type(8)))  float  v8f;

// ---------------------------------------------------------------- helpers ---

// Software RNE f32 -> bf16 bits (only used in the tiny one-shot pack kernels).
__device__ __forceinline__ unsigned bfbits(float f) {
  union { float f; unsigned u; } v; v.f = f;
  unsigned r = v.u + 0x7FFFu + ((v.u >> 16) & 1u);
  return r >> 16;
}

__device__ __forceinline__ float wsum32(float x) {
  #pragma unroll
  for (int o = 16; o > 0; o >>= 1) x += __shfl_xor(x, o, 32);
  return x;
}

// Reduction across the 16 lanes of a half-wave (stays within lanes 0-15 / 16-31).
__device__ __forceinline__ float hsum16(float x) {
  #pragma unroll
  for (int o = 8; o > 0; o >>= 1) x += __shfl_xor(x, o, 32);
  return x;
}

// LayerNorm over a 128-wide row held as 4 values/lane across one wave32.
__device__ __forceinline__ void ln4(float* v, const float* g, const float* b) {
  float s = v[0] + v[1] + v[2] + v[3];
  float mu = wsum32(s) * (1.0f / 128.0f);
  float q = 0.f;
  #pragma unroll
  for (int i = 0; i < 4; ++i) { float d = v[i] - mu; q += d * d; }
  float var = wsum32(q) * (1.0f / 128.0f);
  float inv = 1.0f / sqrtf(var + 1e-5f);
  #pragma unroll
  for (int i = 0; i < 4; ++i) v[i] = (v[i] - mu) * inv * g[i] + b[i];
}

union BFrag { uint4 q[2]; v16bf v; };

// A-fragment (16-bit A 16x32 layout) from 16 f32 via native bf16 cvt.
__device__ __forceinline__ v16bf make_afrag(float4 a0, float4 a1, float4 a2, float4 a3) {
  float fa[16] = { a0.x, a0.y, a0.z, a0.w, a1.x, a1.y, a1.z, a1.w,
                   a2.x, a2.y, a2.z, a2.w, a3.x, a3.y, a3.z, a3.w };
  v16bf af;
  #pragma unroll
  for (int i = 0; i < 16; ++i) af[i] = (__bf16)fa[i];
  return af;
}

// ------------------------------------------------------------ weight pack ---
// W[K,128] f32 row-major -> Wpk[n][k/2] = {bf16(W[k][n]), bf16(W[k+1][n])}:
// a WMMA B-fragment becomes two 4-dword runs per lane, 8 dwords apart.
__global__ void pack_w(const float* __restrict__ W, unsigned* __restrict__ out, int K) {
  const int idx = blockIdx.x * blockDim.x + threadIdx.x;   // 128 * (K/2) entries
  const int kp2 = K >> 1;
  if (idx >= 128 * kp2) return;
  const int n  = idx / kp2;
  const int kp = idx - n * kp2;
  const unsigned lo = bfbits(W[(size_t)(2 * kp)     * H + n]);
  const unsigned hi = bfbits(W[(size_t)(2 * kp + 1) * H + n]);
  out[idx] = lo | (hi << 16);
}

// ------------------------------------------------------------ GEMM (WMMA) ---
// out[M,128] = epilogue(A[M,K] @ W + bias). One wave -> 16 rows x 128 cols.
// Packed W is staged in LDS once per workgroup (shared by all 8 waves).
// relu: apply ReLU. lng/lnb non-null: row LayerNorm after ReLU (fused embed).
template <int K>
__global__ void wgemm128(const float* __restrict__ A, const unsigned* __restrict__ Wpk,
                         const float* __restrict__ bias, float* __restrict__ out,
                         int M, int relu,
                         const float* __restrict__ lng, const float* __restrict__ lnb) {
  constexpr int kp2 = K >> 1;
  extern __shared__ unsigned lds_w[];            // 128 * kp2 dwords

  // cooperative W stage (uint4-wide), BEFORE any wave can exit
  {
    uint4* dst = (uint4*)lds_w;
    const uint4* src = (const uint4*)Wpk;
    const int nvec = 32 * kp2;                   // 128*kp2/4
    for (int i = threadIdx.x; i < nvec; i += 256) dst[i] = src[i];
  }
  __syncthreads();

  const int wave = threadIdx.x >> 5;
  const int lane = threadIdx.x & 31;
  const int tile = blockIdx.x * 8 + wave;        // wave-uniform
  if (tile * 16 >= M) return;                    // whole wave exits -> EXEC all-1
  const int m    = tile * 16 + (lane & 15);
  const int koff = (lane >> 4) * 8;
  const float* __restrict__ arow = A + (size_t)m * K;
  // per-lane B base in LDS: row (lane&15), k-offset (lane>>4)*8 -> dword (lane>>4)*4
  const unsigned* wcol = lds_w + (lane & 15) * kp2 + (lane >> 4) * 4;

  v8f acc[8];
  #pragma unroll
  for (int t = 0; t < 8; ++t) {
    #pragma unroll
    for (int j = 0; j < 8; ++j) acc[t][j] = 0.0f;
  }

  #pragma unroll
  for (int k0 = 0; k0 < K; k0 += 32) {
    const float4* ap0 = (const float4*)(arow + k0 + koff);       // 32B aligned
    const float4* ap1 = (const float4*)(arow + k0 + 16 + koff);
    const v16bf af = make_afrag(ap0[0], ap0[1], ap1[0], ap1[1]);
    #pragma unroll
    for (int t = 0; t < 8; ++t) {
      const uint4* wp = (const uint4*)(wcol + t * 16 * kp2 + (k0 >> 1));  // LDS, 16B aligned
      BFrag bf; bf.q[0] = wp[0]; bf.q[1] = wp[2];
      acc[t] = __builtin_amdgcn_wmma_f32_16x16x32_bf16(
          false, af, false, bf.v, (short)0, acc[t], false, false);
    }
  }

  const int rr = (lane >> 4) * 8;
  const int cc = lane & 15;

  // bias (+ optional ReLU)
  #pragma unroll
  for (int t = 0; t < 8; ++t) {
    const float bv = bias[t * 16 + cc];
    #pragma unroll
    for (int v = 0; v < 8; ++v) {
      float val = acc[t][v] + bv;
      if (relu) val = fmaxf(val, 0.0f);
      acc[t][v] = val;
    }
  }

  // optional fused row LayerNorm: row (rr+v) spans acc[t][v] over 16 lanes
  if (lng != nullptr) {
    float mu[8], rs[8];
    #pragma unroll
    for (int v = 0; v < 8; ++v) {
      float s = 0.f;
      #pragma unroll
      for (int t = 0; t < 8; ++t) s += acc[t][v];
      mu[v] = hsum16(s) * (1.0f / 128.0f);
    }
    #pragma unroll
    for (int v = 0; v < 8; ++v) {
      float q = 0.f;
      #pragma unroll
      for (int t = 0; t < 8; ++t) { float d = acc[t][v] - mu[v]; q += d * d; }
      rs[v] = 1.0f / sqrtf(hsum16(q) * (1.0f / 128.0f) + 1e-5f);
    }
    #pragma unroll
    for (int t = 0; t < 8; ++t) {
      const float gk = lng[t * 16 + cc];
      const float bk = lnb[t * 16 + cc];
      #pragma unroll
      for (int v = 0; v < 8; ++v)
        acc[t][v] = (acc[t][v] - mu[v]) * rs[v] * gk + bk;
    }
  }

  #pragma unroll
  for (int t = 0; t < 8; ++t) {
    #pragma unroll
    for (int v = 0; v < 8; ++v)
      out[(size_t)(tile * 16 + rr + v) * H + t * 16 + cc] = acc[t][v];
  }
}

// Scorer GEMM: A row m = concat(h[row[m]], h[col[m]], e[m]) (K=384),
// out = relu(A @ s1 + b1) -> [E,128]. s1 pre-packed bf16, staged in LDS (96 KB).
__global__ void wgemm_feat(const float* __restrict__ h, const float* __restrict__ e,
                           const int* __restrict__ eidx, const unsigned* __restrict__ s1pk,
                           const float* __restrict__ b1, float* __restrict__ out, int E) {
  constexpr int kp2 = 192;   // 384/2
  extern __shared__ unsigned lds_w[];            // 128 * 192 dwords = 96 KB

  {
    uint4* dst = (uint4*)lds_w;
    const uint4* src = (const uint4*)s1pk;
    const int nvec = 32 * kp2;
    for (int i = threadIdx.x; i < nvec; i += 256) dst[i] = src[i];
  }
  __syncthreads();

  const int wave = threadIdx.x >> 5;
  const int lane = threadIdx.x & 31;
  const int tile = blockIdx.x * 8 + wave;
  if (tile * 16 >= E) return;
  const int m    = tile * 16 + (lane & 15);
  const int koff = (lane >> 4) * 8;
  const float* __restrict__ p0 = h + (size_t)eidx[m] * H;       // h[row]
  const float* __restrict__ p1 = h + (size_t)eidx[E + m] * H;   // h[col]
  const float* __restrict__ p2 = e + (size_t)m * H;             // e
  const unsigned* wcol = lds_w + (lane & 15) * kp2 + (lane >> 4) * 4;

  v8f acc[8];
  #pragma unroll
  for (int t = 0; t < 8; ++t) {
    #pragma unroll
    for (int j = 0; j < 8; ++j) acc[t][j] = 0.0f;
  }

  #pragma unroll
  for (int k0 = 0; k0 < 384; k0 += 32) {
    float4 fr[4];
    #pragma unroll
    for (int half = 0; half < 2; ++half) {
      const int ks  = k0 + half * 16 + koff;       // 8-run stays in one 128-region
      const int reg = ks >> 7;
      const float* p = (reg == 0) ? p0 : ((reg == 1) ? p1 : p2);
      const float4* ap = (const float4*)(p + (ks & 127));
      fr[half * 2]     = ap[0];
      fr[half * 2 + 1] = ap[1];
    }
    const v16bf af = make_afrag(fr[0], fr[1], fr[2], fr[3]);
    #pragma unroll
    for (int t = 0; t < 8; ++t) {
      const uint4* wp = (const uint4*)(wcol + t * 16 * kp2 + (k0 >> 1));  // LDS
      BFrag bf; bf.q[0] = wp[0]; bf.q[1] = wp[2];
      acc[t] = __builtin_amdgcn_wmma_f32_16x16x32_bf16(
          false, af, false, bf.v, (short)0, acc[t], false, false);
    }
  }

  const int rr = (lane >> 4) * 8;
  const int cc = lane & 15;
  #pragma unroll
  for (int t = 0; t < 8; ++t) {
    const float bv = b1[t * 16 + cc];
    #pragma unroll
    for (int v = 0; v < 8; ++v)
      out[(size_t)(tile * 16 + rr + v) * H + t * 16 + cc] = fmaxf(acc[t][v] + bv, 0.0f);
  }
}

// ------------------------------------------------------- elementwise passes ---

__global__ void zero_f32(float* __restrict__ p, size_t n) {
  size_t i = (size_t)blockIdx.x * blockDim.x + threadIdx.x;
  if (i < n) p[i] = 0.0f;
}

// Fused edge pass: e_ji / e_ik, residual+LN, sigmoids, feature-sum norm,
// both messages scatter-added into hagg, e updated in place to e_ji.
__global__ void edge_update(float* __restrict__ e, const float* __restrict__ B1e,
                            const float* __restrict__ A2h, const float* __restrict__ A3h,
                            const float* __restrict__ B2h, const float* __restrict__ B3h,
                            const int* __restrict__ eidx,
                            const float* __restrict__ lng, const float* __restrict__ lnb,
                            float* __restrict__ hagg, int E) {
  const int gw = (int)(((size_t)blockIdx.x * blockDim.x + threadIdx.x) >> 5);
  if (gw >= E) return;
  const int lane = threadIdx.x & 31;
  const int r = eidx[gw], c = eidx[E + gw];
  const size_t eb = (size_t)gw * H;
  const size_t rb = (size_t)r * H, cb = (size_t)c * H;

  float ev[4], nji[4], nik[4], gk[4], bk[4], a2[4], a3[4];
  #pragma unroll
  for (int i = 0; i < 4; ++i) {
    const int d = lane * 4 + i;
    const float v1  = B1e[eb + d];
    const float b2r = B2h[rb + d], b2c = B2h[cb + d];
    const float b3r = B3h[rb + d], b3c = B3h[cb + d];
    ev[i]  = e[eb + d];
    nji[i] = fmaxf(v1 + b2r + b3c, 0.0f);
    nik[i] = fmaxf(v1 + b2c + b3r, 0.0f);
    gk[i] = lng[d]; bk[i] = lnb[d];
    a2[i] = A2h[rb + d];           // A2h[row]
    a3[i] = A3h[cb + d];           // A3h[col]
  }
  ln4(nji, gk, bk);
  ln4(nik, gk, bk);

  float eji[4], sji[4], sik[4];
  float ssji = 0.f, ssik = 0.f;
  #pragma unroll
  for (int i = 0; i < 4; ++i) {
    eji[i] = ev[i] + nji[i];
    const float eik = ev[i] + nik[i];
    sji[i] = 1.0f / (1.0f + expf(-eji[i]));
    sik[i] = 1.0f / (1.0f + expf(-eik));
    ssji += sji[i]; ssik += sik[i];
  }
  const float dji = 1.0f / (wsum32(ssji) + 1e-6f);
  const float dik = 1.0f / (wsum32(ssik) + 1e-6f);

  #pragma unroll
  for (int i = 0; i < 4; ++i) {
    const int d = lane * 4 + i;
    atomicAdd(hagg + cb + d, a2[i] * sji[i] * dji);   // segment_sum(m_ji, col)
    atomicAdd(hagg + rb + d, a3[i] * sik[i] * dik);   // segment_sum(m_ik, row)
    e[eb + d] = eji[i];                               // new e = e_ji
  }
}

// h = h + LN(relu(A1h + hagg)); one wave per node.
__global__ void node_update(float* __restrict__ h, const float* __restrict__ A1h,
                            const float* __restrict__ hagg,
                            const float* __restrict__ g, const float* __restrict__ b,
                            int N) {
  const int gw = (int)(((size_t)blockIdx.x * blockDim.x + threadIdx.x) >> 5);
  if (gw >= N) return;
  const int lane = threadIdx.x & 31;
  const size_t base = (size_t)gw * H;
  float t[4], gk[4], bk[4], hv[4];
  #pragma unroll
  for (int i = 0; i < 4; ++i) {
    const int d = lane * 4 + i;
    t[i]  = fmaxf(A1h[base + d] + hagg[base + d], 0.0f);
    gk[i] = g[d]; bk[i] = b[d];
    hv[i] = h[base + d];
  }
  ln4(t, gk, bk);
  #pragma unroll
  for (int i = 0; i < 4; ++i) h[base + lane * 4 + i] = hv[i] + t[i];
}

// score[m] = relu_feat[m] . s2 + s2_b; one wave per edge.
__global__ void score_kernel(const float* __restrict__ rf, const float* __restrict__ s2w,
                             const float* __restrict__ s2b, float* __restrict__ out, int E) {
  const int gw = (int)(((size_t)blockIdx.x * blockDim.x + threadIdx.x) >> 5);
  if (gw >= E) return;
  const int lane = threadIdx.x & 31;
  float s = 0.f;
  #pragma unroll
  for (int i = 0; i < 4; ++i) {
    const int d = lane * 4 + i;
    s += rf[(size_t)gw * H + d] * s2w[d];
  }
  s = wsum32(s);
  if (lane == 0) out[gw] = s + s2b[0];
}

// ------------------------------------------------------------------ driver ---

extern "C" void kernel_launch(void* const* d_in, const int* in_sizes, int n_in,
                              void* d_out, int out_size, void* d_ws, size_t ws_size,
                              hipStream_t stream) {
  const float* x         = (const float*)d_in[0];
  const float* edge_attr = (const float*)d_in[1];
  const int*   eidx      = (const int*)  d_in[2];
  const float* W11w = (const float*)d_in[3];  const float* W11b = (const float*)d_in[4];
  const float* W12w = (const float*)d_in[5];  const float* W12b = (const float*)d_in[6];
  const float* W21w = (const float*)d_in[7];  const float* W21b = (const float*)d_in[8];
  const float* W22w = (const float*)d_in[9];  const float* W22b = (const float*)d_in[10];
  const float* ln1g = (const float*)d_in[11]; const float* ln1b = (const float*)d_in[12];
  const float* ln2g = (const float*)d_in[13]; const float* ln2b = (const float*)d_in[14];
  const float* Aw   = (const float*)d_in[15]; const float* Ab   = (const float*)d_in[16];
  const float* Bw   = (const float*)d_in[17]; const float* Bb   = (const float*)d_in[18];
  const float* lnhg = (const float*)d_in[19]; const float* lnhb = (const float*)d_in[20];
  const float* lneg = (const float*)d_in[21]; const float* lneb = (const float*)d_in[22];
  const float* s1w  = (const float*)d_in[23]; const float* s1b  = (const float*)d_in[24];
  const float* s2w  = (const float*)d_in[25]; const float* s2b  = (const float*)d_in[26];
  float* out = (float*)d_out;

  const int N = in_sizes[0] / 64;   // FN = 64
  const int E = in_sizes[1] / 32;   // FE = 32

  // ---- workspace layout (f32 buffers ~1.0 GB, then packed bf16 weights) ----
  float* ws = (float*)d_ws;
  const size_t nh = (size_t)N * H;
  const size_t eh = (size_t)E * H;
  float* h    = ws;
  float* A1h  = h    + nh;
  float* A2h  = A1h  + nh;
  float* A3h  = A2h  + nh;
  float* B2h  = A3h  + nh;
  float* B3h  = B2h  + nh;
  float* hagg = B3h  + nh;
  float* e    = hagg + nh;
  float* B1e  = e    + eh;   // also reused as edge-embed tmp and relu_feat

  unsigned* pkpool = (unsigned*)(B1e + eh);
  unsigned* pW11 = pkpool;             // 64*K uints each
  unsigned* pW12 = pW11 + 64 * 64;
  unsigned* pW21 = pW12 + 64 * 128;
  unsigned* pW22 = pW21 + 64 * 32;
  unsigned* pAw  = pW22 + 64 * 128;    // 9 matrices of 64*128
  unsigned* pBw  = pAw  + 9 * 64 * 128;
  unsigned* pS1  = pBw  + 9 * 64 * 128;

  const dim3 blk(256);
  auto packBlocks = [](int K) { return (128 * (K / 2) + 255) / 256; };

  // ---- pre-pack all weights to bf16 B-fragment layout ----
  pack_w<<<packBlocks(64),  blk, 0, stream>>>(W11w, pW11, 64);
  pack_w<<<packBlocks(128), blk, 0, stream>>>(W12w, pW12, 128);
  pack_w<<<packBlocks(32),  blk, 0, stream>>>(W21w, pW21, 32);
  pack_w<<<packBlocks(128), blk, 0, stream>>>(W22w, pW22, 128);
  for (int j = 0; j < 9; ++j) {
    pack_w<<<packBlocks(128), blk, 0, stream>>>(Aw + (size_t)j * H * H, pAw + (size_t)j * 64 * 128, 128);
    pack_w<<<packBlocks(128), blk, 0, stream>>>(Bw + (size_t)j * H * H, pBw + (size_t)j * 64 * 128, 128);
  }
  pack_w<<<packBlocks(384), blk, 0, stream>>>(s1w, pS1, 384);

  const int nodeTiles = (N + 15) / 16, nodeBlocks = (nodeTiles + 7) / 8;
  const int edgeTiles = (E + 15) / 16, edgeBlocks = (edgeTiles + 7) / 8;
  const int nodeWaveBlocks = (N + 7) / 8;
  const int edgeWaveBlocks = (E + 7) / 8;
  const int zeroBlocks = (int)((nh + 255) / 256);

  // dynamic-LDS bytes for each GEMM flavor (128 * K/2 dwords)
  const size_t lds32  = 128 * 16  * 4;   //  8 KB
  const size_t lds64  = 128 * 32  * 4;   // 16 KB
  const size_t lds128 = 128 * 64  * 4;   // 32 KB
  const size_t lds384 = 128 * 192 * 4;   // 96 KB

  // --- input embeddings (GEMM + fused ReLU+LN, then projection GEMM) ---
  wgemm128<64><<<nodeBlocks, blk, lds64, stream>>>(x, pW11, W11b, A1h, N, 1, ln1g, ln1b);
  wgemm128<128><<<nodeBlocks, blk, lds128, stream>>>(A1h, pW12, W12b, h, N, 0, nullptr, nullptr);

  wgemm128<32><<<edgeBlocks, blk, lds32, stream>>>(edge_attr, pW21, W21b, B1e, E, 1, ln2g, ln2b);
  wgemm128<128><<<edgeBlocks, blk, lds128, stream>>>(B1e, pW22, W22b, e, E, 0, nullptr, nullptr);

  // --- gated layers ---
  for (int l = 0; l < 3; ++l) {
    const unsigned* pAl = pAw + (size_t)l * 3 * 64 * 128;
    const unsigned* pBl = pBw + (size_t)l * 3 * 64 * 128;
    const float* Abl = Ab + (size_t)l * 3 * H;
    const float* Bbl = Bb + (size_t)l * 3 * H;

    wgemm128<128><<<nodeBlocks, blk, lds128, stream>>>(h, pAl + 0 * 64 * 128, Abl + 0 * H, A1h, N, 0, nullptr, nullptr);
    wgemm128<128><<<nodeBlocks, blk, lds128, stream>>>(h, pAl + 1 * 64 * 128, Abl + 1 * H, A2h, N, 0, nullptr, nullptr);
    wgemm128<128><<<nodeBlocks, blk, lds128, stream>>>(h, pAl + 2 * 64 * 128, Abl + 2 * H, A3h, N, 0, nullptr, nullptr);
    wgemm128<128><<<nodeBlocks, blk, lds128, stream>>>(h, pBl + 1 * 64 * 128, Bbl + 1 * H, B2h, N, 0, nullptr, nullptr);
    wgemm128<128><<<nodeBlocks, blk, lds128, stream>>>(h, pBl + 2 * 64 * 128, Bbl + 2 * H, B3h, N, 0, nullptr, nullptr);
    wgemm128<128><<<edgeBlocks, blk, lds128, stream>>>(e, pBl + 0 * 64 * 128, Bbl + 0 * H, B1e, E, 0, nullptr, nullptr);

    zero_f32<<<zeroBlocks, blk, 0, stream>>>(hagg, nh);
    edge_update<<<edgeWaveBlocks, blk, 0, stream>>>(e, B1e, A2h, A3h, B2h, B3h, eidx,
                                                    lneg + (size_t)l * H, lneb + (size_t)l * H,
                                                    hagg, E);
    node_update<<<nodeWaveBlocks, blk, 0, stream>>>(h, A1h, hagg,
                                                    lnhg + (size_t)l * H, lnhb + (size_t)l * H, N);
  }

  // --- scorer ---
  wgemm_feat<<<edgeBlocks, blk, lds384, stream>>>(h, e, eidx, pS1, s1b, B1e, E);
  score_kernel<<<edgeWaveBlocks, blk, 0, stream>>>(B1e, s2w, s2b, out, E);
}
